// AutoregGPT2Model_42777874268563
// MI455X (gfx1250) — matrix operand order repaired and checked by reading, not compiled, not run on hardware
//
#include <hip/hip_runtime.h>

typedef __bf16 bf16;
typedef bf16  v16bf __attribute__((ext_vector_type(16)));
typedef bf16  v8bf  __attribute__((ext_vector_type(8)));
typedef float v8f   __attribute__((ext_vector_type(8)));

#define DM    1024
#define LMAX  288
#define SMAXC 288
#define NB    16
#define NPROP 512
#define NSTEP 32

// ---------------------------------------------------------------------------
// WMMA GEMM:  out[M,N] = alpha * (A[M,K] @ B[K,N]) (+bias) (+res)
// B supplied TRANSPOSED as BT[N,K] row-major -> contiguous per-lane loads
// with element s at k = 16*half + s (same order as the A fragment).
// One wave computes a 16 x (16*NJ) tile. Edge tiles CLAMP row/col indices so
// every load is unconditional & in-bounds (no exec-mask churn in the K-loop);
// garbage lanes are discarded by the store-side guards.
// ---------------------------------------------------------------------------
template<int NJ, bool STORE_BF16>
__global__ __launch_bounds__(32)
void gemm_bt_wmma(const bf16* __restrict__ A,  size_t strideA, int lda,
                  const bf16* __restrict__ BT, size_t strideB, int ldb,
                  const float* __restrict__ bias,
                  const float* __restrict__ res, size_t strideR, int ldr,
                  void* __restrict__ outp, size_t strideO, int ldo,
                  int M, int N, int K, float alpha)
{
    const int lane = threadIdx.x & 31;
    const int half = lane >> 4;      // 0: lanes 0-15, 1: lanes 16-31
    const int l15  = lane & 15;
    const int tile_m = blockIdx.y * 16;
    const int tile_n = blockIdx.x * (16 * NJ);
    const int bz = blockIdx.z;

    A  += (size_t)bz * strideA;
    BT += (size_t)bz * strideB;
    const float* resb = res ? res + (size_t)bz * strideR : nullptr;
    float* outF = (float*)outp + (STORE_BF16 ? 0 : (size_t)bz * strideO);
    bf16*  outH = (bf16*)outp  + (STORE_BF16 ? (size_t)bz * strideO : 0);

    v8f acc[NJ];
#pragma unroll
    for (int j = 0; j < NJ; ++j)
#pragma unroll
        for (int r = 0; r < 8; ++r) acc[j][r] = 0.0f;

    // clamp (edge-tile lanes produce garbage that is never stored)
    int arow = tile_m + l15;                 // lanes 0-15 & 16-31 both hold M=0..15
    arow = (arow < M) ? arow : (M - 1);
    const bf16* Arow = A + (size_t)arow * lda;

    const bf16* Brow[NJ];
#pragma unroll
    for (int j = 0; j < NJ; ++j) {
        int ncol = tile_n + j * 16 + l15;
        ncol = (ncol < N) ? ncol : (N - 1);
        Brow[j] = BT + (size_t)ncol * ldb + half * 16;
    }

    union { v16bf v; v8bf h[2]; } au;
#pragma unroll 2
    for (int kk = 0; kk < K; kk += 32) {
        // A frag: elems 0..7 = K kk+half*16.. ; elems 8..15 = K kk+half*16+8..
        au.h[0] = *(const v8bf*)(Arow + kk + half * 16);
        au.h[1] = *(const v8bf*)(Arow + kk + half * 16 + 8);
        const v16bf afrag = au.v;
#pragma unroll
        for (int j = 0; j < NJ; ++j) {
            const v16bf bfrag = *(const v16bf*)(Brow[j] + kk);
            acc[j] = __builtin_amdgcn_wmma_f32_16x16x32_bf16(
                false, afrag, false, bfrag, (short)0, acc[j], false, false);
        }
    }

    // C/D layout: VGPR r -> row r (lanes 0-15) / row r+8 (lanes 16-31), col = l15
#pragma unroll
    for (int j = 0; j < NJ; ++j) {
        const int col = tile_n + j * 16 + l15;
        if (col >= N) continue;
        const float bv = bias ? bias[col] : 0.0f;
#pragma unroll
        for (int r = 0; r < 8; ++r) {
            const int row = tile_m + r + half * 8;
            if (row >= M) continue;
            float v = acc[j][r] * alpha + bv;
            if (STORE_BF16) {
                outH[(size_t)row * ldo + col] = (bf16)v;
            } else {
                if (resb) v += resb[(size_t)row * ldr + col];
                outF[(size_t)row * ldo + col] = v;
            }
        }
    }
}

// ---------------------------------------------------------------------------
// LayerNorm over D=1024 (f32 in -> bf16 out). grid = (rows, batch)
// ---------------------------------------------------------------------------
__global__ __launch_bounds__(256)
void ln_kernel(const float* __restrict__ x, size_t xb,
               bf16* __restrict__ y, size_t yb,
               const float* __restrict__ g, const float* __restrict__ be)
{
    __shared__ float s1[256], s2[256];
    const int s = blockIdx.x, b = blockIdx.y, t = threadIdx.x;
    const float* xr = x + (size_t)b * xb + (size_t)s * DM;
    bf16* yr = y + (size_t)b * yb + (size_t)s * DM;
    float sum = 0.f, sq = 0.f;
    for (int i = t; i < DM; i += 256) { float v = xr[i]; sum += v; sq += v * v; }
    s1[t] = sum; s2[t] = sq; __syncthreads();
    for (int w = 128; w > 0; w >>= 1) {
        if (t < w) { s1[t] += s1[t + w]; s2[t] += s2[t + w]; }
        __syncthreads();
    }
    const float mu  = s1[0] * (1.0f / DM);
    const float var = s2[0] * (1.0f / DM) - mu * mu;
    const float inv = rsqrtf(var + 1e-5f);
    for (int i = t; i < DM; i += 256)
        yr[i] = (bf16)((xr[i] - mu) * inv * g[i] + be[i]);
}

// ---------------------------------------------------------------------------
// Causal softmax row q: cols 0..q active, rest written as 0 (pads K-loop)
// ---------------------------------------------------------------------------
__global__ __launch_bounds__(256)
void softmax_kernel(const float* __restrict__ sc, bf16* __restrict__ at)
{
    __shared__ float red[256];
    const int q = blockIdx.x, b = blockIdx.y, t = threadIdx.x;
    const float* row = sc + ((size_t)b * SMAXC + q) * SMAXC;
    bf16* orow = at + ((size_t)b * SMAXC + q) * SMAXC;
    const int n = q + 1;
    float m = -1e30f;
    for (int i = t; i < n; i += 256) m = fmaxf(m, row[i]);
    red[t] = m; __syncthreads();
    for (int w = 128; w > 0; w >>= 1) {
        if (t < w) red[t] = fmaxf(red[t], red[t + w]);
        __syncthreads();
    }
    m = red[0]; __syncthreads();
    float ssum = 0.f;
    for (int i = t; i < n; i += 256) ssum += __expf(row[i] - m);
    red[t] = ssum; __syncthreads();
    for (int w = 128; w > 0; w >>= 1) {
        if (t < w) red[t] += red[t + w];
        __syncthreads();
    }
    const float invs = 1.0f / red[0];
    for (int i = t; i < SMAXC; i += 256)
        orow[i] = (bf16)((i < n) ? __expf(row[i] - m) * invs : 0.0f);
}

// gelu_new in-place on bf16 fc activations
__global__ void gelu_kernel(bf16* __restrict__ fc, int S, size_t bstride)
{
    const int b = blockIdx.y;
    const int idx = blockIdx.x * 256 + threadIdx.x;
    if (idx >= S * 4096) return;
    bf16* p = fc + (size_t)b * bstride + idx;
    const float x = (float)*p;
    const float v = 0.5f * x * (1.0f + tanhf(0.7978845608028654f * (x + 0.044715f * x * x * x)));
    *p = (bf16)v;
}

// v^T: vT[b][d][s] = qkv[b][s][2D+d], zero-padded for s in [S, 288)
__global__ void vtrans_kernel(const bf16* __restrict__ qkv, bf16* __restrict__ vT, int S)
{
    const int b = blockIdx.y;
    const int idx = blockIdx.x * 256 + threadIdx.x;
    if (idx >= DM * SMAXC) return;
    const int d = idx / SMAXC, s = idx % SMAXC;
    vT[((size_t)b * DM + d) * SMAXC + s] =
        (s < S) ? qkv[((size_t)b * LMAX + s) * 3072 + 2048 + d] : (bf16)0.0f;
}

// logits -> d_out + append hard-thresholded successor token row
__global__ __launch_bounds__(512)
void append_kernel(const float* __restrict__ logits, float* __restrict__ out,
                   float* __restrict__ all_toks, int step)
{
    const int b = blockIdx.x, t = threadIdx.x; // 512 threads = NPROP
    const float lg = logits[b * NPROP + t];
    out[((size_t)b * NSTEP + step) * NPROP + t] = lg;
    float* row = all_toks + ((size_t)b * LMAX + (256 + step)) * DM;
    row[t] = 0.0f;
    row[NPROP + t] = (lg > 0.0f) ? 1.0f : 0.0f;
}

// weight f32 [K,N] -> bf16 transposed [N,K]
__global__ void cvt_transpose(const float* __restrict__ W, bf16* __restrict__ BT,
                              int K, int N)
{
    const int idx = blockIdx.x * 256 + threadIdx.x;
    if (idx >= N * K) return;
    const int n = idx / K, k = idx % K;
    BT[idx] = (bf16)W[(size_t)k * N + n];
}

__global__ void cvt_kernel(const float* __restrict__ W, bf16* __restrict__ O, int n)
{
    const int idx = blockIdx.x * 256 + threadIdx.x;
    if (idx < n) O[idx] = (bf16)W[idx];
}

// tokens (0/1 int) -> f32 embeddings in all_toks[:, 0:256, :]
__global__ void tokens_kernel(const int* __restrict__ tok, float* __restrict__ all_toks)
{
    const int idx = blockIdx.x * 256 + threadIdx.x; // B*256*D total
    const int b = idx / (256 * DM);
    const int r = idx % (256 * DM);
    const int s = r / DM, d = r % DM;
    all_toks[((size_t)b * LMAX + s) * DM + d] = (float)tok[idx];
}

// ---------------------------------------------------------------------------
extern "C" void kernel_launch(void* const* d_in, const int* in_sizes, int n_in,
                              void* d_out, int out_size, void* d_ws, size_t ws_size,
                              hipStream_t stream)
{
    (void)in_sizes; (void)n_in; (void)out_size; (void)ws_size;
    const int*   tokens  = (const int*)  d_in[0];
    const float* W_qkv   = (const float*)d_in[1];
    const float* b_qkv   = (const float*)d_in[2];
    const float* W_aproj = (const float*)d_in[3];
    const float* b_aproj = (const float*)d_in[4];
    const float* g_ln1   = (const float*)d_in[5];
    const float* b_ln1   = (const float*)d_in[6];
    const float* g_ln2   = (const float*)d_in[7];
    const float* b_ln2   = (const float*)d_in[8];
    const float* W_fc    = (const float*)d_in[9];
    const float* b_fc    = (const float*)d_in[10];
    const float* W_mproj = (const float*)d_in[11];
    const float* b_mproj = (const float*)d_in[12];
    const float* g_lnf   = (const float*)d_in[13];
    const float* b_lnf   = (const float*)d_in[14];
    const float* W_score = (const float*)d_in[15];

    // --- workspace carve-up (everything fits in the 192MB L2) ---
    size_t off = 0;
    auto alloc = [&](size_t bytes) -> void* {
        off = (off + 255) & ~(size_t)255;
        void* p = (char*)d_ws + off;
        off += bytes;
        return p;
    };
    bf16*  wqkvT   = (bf16*) alloc((size_t)3072 * 1024 * 2);
    bf16*  waprojT = (bf16*) alloc((size_t)1024 * 1024 * 2);
    bf16*  wfcT    = (bf16*) alloc((size_t)4096 * 1024 * 2);
    bf16*  wmprojT = (bf16*) alloc((size_t)1024 * 4096 * 2);
    bf16*  wscoreB = (bf16*) alloc((size_t)NPROP * 1024 * 2);
    float* all_toks= (float*)alloc((size_t)NB * LMAX * DM * 4);
    float* x_work  = (float*)alloc((size_t)NB * LMAX * DM * 4);
    bf16*  h_bf    = (bf16*) alloc((size_t)NB * LMAX * DM * 2);
    bf16*  qkv_bf  = (bf16*) alloc((size_t)NB * LMAX * 3072 * 2);
    float* scores  = (float*)alloc((size_t)NB * SMAXC * SMAXC * 4);
    bf16*  attn_bf = (bf16*) alloc((size_t)NB * SMAXC * SMAXC * 2);
    bf16*  vT      = (bf16*) alloc((size_t)NB * DM * SMAXC * 2);
    bf16*  attnout = (bf16*) alloc((size_t)NB * LMAX * DM * 2);
    bf16*  fc_bf   = (bf16*) alloc((size_t)NB * LMAX * 4096 * 2);
    bf16*  lnf_bf  = (bf16*) alloc((size_t)NB * DM * 2);
    float* logits  = (float*)alloc((size_t)NB * NPROP * 4);

    const size_t bsD  = (size_t)LMAX * DM;      // batch stride, D-wide acts
    const size_t bs3D = (size_t)LMAX * 3072;
    const size_t bsS  = (size_t)SMAXC * SMAXC;
    const size_t bsVT = (size_t)DM * SMAXC;
    const int MALL = NB * LMAX;                 // flattened rows (mult of 16)

    // --- one-time (per launch) weight conversion + token embedding ---
    cvt_transpose<<<(3072 * 1024 + 255) / 256, 256, 0, stream>>>(W_qkv,   wqkvT,   1024, 3072);
    cvt_transpose<<<(1024 * 1024 + 255) / 256, 256, 0, stream>>>(W_aproj, waprojT, 1024, 1024);
    cvt_transpose<<<(4096 * 1024 + 255) / 256, 256, 0, stream>>>(W_fc,    wfcT,    1024, 4096);
    cvt_transpose<<<(4096 * 1024 + 255) / 256, 256, 0, stream>>>(W_mproj, wmprojT, 4096, 1024);
    cvt_kernel<<<(NPROP * 1024 + 255) / 256, 256, 0, stream>>>(W_score, wscoreB, NPROP * 1024);
    tokens_kernel<<<(NB * 256 * DM) / 256, 256, 0, stream>>>(tokens, all_toks);

    for (int step = 0; step < NSTEP; ++step) {
        const int S = 256 + step;

        // h = LN1(x)
        ln_kernel<<<dim3(S, NB), 256, 0, stream>>>(all_toks, bsD, h_bf, bsD, g_ln1, b_ln1);

        // qkv = h @ W_qkv + b_qkv   (bf16 out), 16x128 tiles
        gemm_bt_wmma<8, true><<<dim3(3072 / 128, MALL / 16, 1), 32, 0, stream>>>(
            h_bf, 0, DM, wqkvT, 0, DM, b_qkv, nullptr, 0, 0,
            qkv_bf, 0, 3072, MALL, 3072, DM, 1.0f);

        // scores = q @ k^T / 32   (per batch, ragged -> 16x64 tiles)
        gemm_bt_wmma<4, false><<<dim3((S + 63) / 64, (S + 15) / 16, NB), 32, 0, stream>>>(
            qkv_bf, bs3D, 3072, qkv_bf + DM, bs3D, 3072, nullptr, nullptr, 0, 0,
            scores, bsS, SMAXC, S, S, DM, 0.03125f);

        // causal softmax -> attn (bf16, zero-padded to 288 cols)
        softmax_kernel<<<dim3(S, NB), 256, 0, stream>>>(scores, attn_bf);

        // v^T for contiguous B-operand loads
        vtrans_kernel<<<dim3((DM * SMAXC + 255) / 256, NB), 256, 0, stream>>>(qkv_bf, vT, S);

        // attnout = attn @ v   (per batch, K = 288 padded)
        gemm_bt_wmma<8, true><<<dim3(DM / 128, (S + 15) / 16, NB), 32, 0, stream>>>(
            attn_bf, bsS, SMAXC, vT, bsVT, SMAXC, nullptr, nullptr, 0, 0,
            attnout, bsD, DM, S, DM, SMAXC, 1.0f);

        // x_work = all_toks + attnout @ W_aproj + b_aproj
        gemm_bt_wmma<8, false><<<dim3(DM / 128, MALL / 16, 1), 32, 0, stream>>>(
            attnout, 0, DM, waprojT, 0, DM, b_aproj, all_toks, 0, DM,
            x_work, 0, DM, MALL, DM, DM, 1.0f);

        // h = LN2(x_work)
        ln_kernel<<<dim3(S, NB), 256, 0, stream>>>(x_work, bsD, h_bf, bsD, g_ln2, b_ln2);

        // fc = h @ W_fc + b_fc (bf16), then gelu in place
        gemm_bt_wmma<8, true><<<dim3(4096 / 128, MALL / 16, 1), 32, 0, stream>>>(
            h_bf, 0, DM, wfcT, 0, DM, b_fc, nullptr, 0, 0,
            fc_bf, 0, 4096, MALL, 4096, DM, 1.0f);
        gelu_kernel<<<dim3((S * 4096 + 255) / 256, NB), 256, 0, stream>>>(
            fc_bf, S, (size_t)LMAX * 4096);

        // x_work += gelu(fc) @ W_mproj + b_mproj
        gemm_bt_wmma<8, false><<<dim3(DM / 128, MALL / 16, 1), 32, 0, stream>>>(
            fc_bf, 0, 4096, wmprojT, 0, 4096, b_mproj, x_work, 0, DM,
            x_work, 0, DM, MALL, DM, 4096, 1.0f);

        // lnf on last token only
        ln_kernel<<<dim3(1, NB), 256, 0, stream>>>(
            x_work + (size_t)(S - 1) * DM, bsD, lnf_bf, (size_t)DM, g_lnf, b_lnf);

        // logits = lnf @ W_score^T  (W_score already [N,K])
        gemm_bt_wmma<8, false><<<dim3(NPROP / 128, 1, 1), 32, 0, stream>>>(
            lnf_bf, 0, DM, wscoreB, 0, DM, nullptr, nullptr, 0, 0,
            logits, 0, NPROP, NB, NPROP, DM, 1.0f);

        // emit logits + append thresholded token row for next step
        append_kernel<<<dim3(NB), 512, 0, stream>>>(logits, (float*)d_out, all_toks, step);
    }
}
